// SimpleRNN_90417651516280
// MI455X (gfx1250) — matrix operand order repaired and checked by reading, not compiled, not run on hardware
//
#include <hip/hip_runtime.h>
#include <hip/hip_bf16.h>
#include <math.h>

// ---------------------------------------------------------------------------
// SimpleRNN on MI455X (gfx1250, wave32, WMMA).
//
// Effective computation (dead code removed from reference):
//   h0[t]  = tanh(x[t] @ Wx[0] + h0[t-1] @ Wh[0] + b[0])     (t = 0..63)
//   h5     = tanh(x[63] @ Wx[5] + h0[63] @ Wh[5] + b[5])
//   out    = sigmoid(h5 @ Wout + bout)                        ([64] floats)
//
// bf16 WMMA (16x16x32) with fp32 accumulation; persistent grid of 64 WGs x
// 256 threads; double-buffered h state; atomic-counter device barrier between
// timesteps. K loops fully unrolled (immediate-offset loads) and software-
// pipelined. The big GEMM's per-lane offsets are laundered through an empty
// asm as 32-bit integers: blocks cross-timestep LICM of B fragments (round-2
// spill bug) while keeping the bases as kernel-arg global pointers so loads
// stay GLOBAL, not FLAT (round-3 flat-load bug).
// ---------------------------------------------------------------------------

typedef __bf16 bf16_t;
typedef __attribute__((ext_vector_type(16))) __bf16 v16bf;
typedef __attribute__((ext_vector_type(8)))  __bf16 v8bf;
typedef __attribute__((ext_vector_type(8)))  float  v8f;

#define UNITS 2048
#define NBATCH 64
#define SEQT  64
#define EMBD  80
#define KXP   96           // EMB padded to multiple of 32
#define NBLK  64           // persistent workgroups (32 output cols each)
#define HBUF  ((size_t)NBATCH * UNITS)   // elements per h buffer

// ---------------------------------------------------------------------------
// One 16x16 C-tile GEMM: C += A[16 x K] * B[K x 16], bf16 in, f32 acc.
// A row-major (lda), B pre-transposed W_t[n][k] (ldb). Fragment layouts used here
// (contiguous k, not the ISA's table order; wave32):
//   A: lane&15 = row, halves j -> K = 16*(lane>>4) + j (contiguous)
//   B: lane&15 = column, halves j -> K = 16*(lane>>4) + j (contiguous)
//   C: lane&15 = column, vgpr v -> row = 8*(lane>>4) + v
// OPQ=true: per-lane byte offsets made opaque (blocks LICM, keeps GLOBAL AS).
// Software-pipelined: next fragments are in flight while current WMMA runs.
// ---------------------------------------------------------------------------
template<int K, bool OPQ>
__device__ __forceinline__ v8f wmma_gemm(const bf16_t* __restrict__ Abase, int lda,
                                         const bf16_t* __restrict__ Bbase, int ldb,
                                         v8f c, int lane)
{
    const int lg  = lane >> 4;
    const int l15 = lane & 15;
    unsigned aoff = (unsigned)(l15 * lda + 16 * lg) * 2u;   // byte offset
    unsigned boff = (unsigned)(l15 * ldb + 16 * lg) * 2u;   // byte offset
    if (OPQ) {
        // Opaque lane offsets: loads can't be proven loop-invariant across
        // timesteps (no giant register caching / scratch spills), but the
        // base pointers remain recognizable as GLOBAL address space.
        asm volatile("" : "+v"(aoff), "+v"(boff));
    }
    const bf16_t* __restrict__ ap =
        (const bf16_t*)((const char*)Abase + aoff);
    const bf16_t* __restrict__ bp =
        (const bf16_t*)((const char*)Bbase + boff);

    // stage 0 preload
    v8bf  alo = *(const v8bf*)(ap);
    v8bf  ahi = *(const v8bf*)(ap + 8);
    v16bf bnx = *(const v16bf*)(bp);
#pragma unroll
    for (int k = 0; k < K; k += 32) {
        v16bf a = __builtin_shufflevector(alo, ahi,
                   0,1,2,3,4,5,6,7,8,9,10,11,12,13,14,15);
        v16bf b = bnx;
        if (k + 32 < K) {              // preload next iteration (folds away)
            alo = *(const v8bf*)(ap + k + 32);
            ahi = *(const v8bf*)(ap + k + 40);
            bnx = *(const v16bf*)(bp + k + 32);
        }
        if ((k & 255) == 0 && (k + 512) < K) {
            __builtin_prefetch(bp + k + 512, 0, 3);   // global_prefetch_b8
        }
        c = __builtin_amdgcn_wmma_f32_16x16x32_bf16(
                /*neg_a=*/false, a, /*neg_b=*/false, b,
                /*c_mod=*/(short)0, c, /*reuse_a=*/false, /*reuse_b=*/false);
    }
    return c;
}

// ---------------------------------------------------------------------------
// Prep kernel 1: transpose+convert Wh[0] and Wh[5] (f32 [2048][2048], k-major)
// into bf16 [n][k] (K-contiguous per output column).
// ---------------------------------------------------------------------------
__global__ void k_transpose_wh(const float* __restrict__ Wh,
                               bf16_t* __restrict__ Wh0t,
                               bf16_t* __restrict__ Wh5t)
{
    unsigned idx = blockIdx.x * 256u + threadIdx.x;     // 2 * 2048 * 2048 total
    if (idx >= 2u * UNITS * UNITS) return;
    unsigned which = idx >> 22;                          // 0 -> layer0, 1 -> layer5
    unsigned r = idx & ((1u << 22) - 1u);
    unsigned n = r >> 11;                                // output column
    unsigned k = r & 2047u;                              // input unit
    const float* src = Wh + (which ? (size_t)5 * UNITS * UNITS : 0);
    bf16_t* dst = which ? Wh5t : Wh0t;
    dst[(size_t)n * UNITS + k] = (bf16_t)src[(size_t)k * UNITS + n];
}

// ---------------------------------------------------------------------------
// Prep kernel 2: Wx[0]/Wx[5] -> bf16 [n][96] (zero-padded K), embedding gather
// -> bf16 x[t][b][96], zero h buffer 0 / output accumulator / barrier counter.
// ---------------------------------------------------------------------------
__global__ void k_prep_misc(const float* __restrict__ Wx,
                            const float* __restrict__ emb,
                            const int*   __restrict__ inputs,
                            bf16_t* __restrict__ Wx0t, bf16_t* __restrict__ Wx5t,
                            bf16_t* __restrict__ xbf,  bf16_t* __restrict__ h01,
                            float*  __restrict__ out_acc, unsigned* __restrict__ bar)
{
    int idx = blockIdx.x * 256 + threadIdx.x;
    if (idx < 2 * UNITS * KXP) {                               // 393216: Wx transpose
        int which = idx / (UNITS * KXP);
        int r = idx % (UNITS * KXP);
        int n = r / KXP;
        int kk = r % KXP;
        float v = (kk < EMBD)
                ? Wx[(size_t)(which ? 5 : 0) * EMBD * UNITS + (size_t)kk * UNITS + n]
                : 0.0f;
        (which ? Wx5t : Wx0t)[(size_t)n * KXP + kk] = (bf16_t)v;
    } else if (idx < 2 * UNITS * KXP + SEQT * NBATCH * KXP) {  // embedding gather
        int j = idx - 2 * UNITS * KXP;
        int e  = j % KXP;
        int q  = j / KXP;
        int bb = q % NBATCH;
        int t  = q / NBATCH;
        float v = 0.0f;
        if (e < EMBD) {
            int tok = inputs[bb * SEQT + t];                   // inputs[b][t]
            v = emb[(size_t)tok * EMBD + e];
        }
        xbf[j] = (bf16_t)v;
    } else if (idx < 2 * UNITS * KXP + SEQT * NBATCH * KXP + NBATCH * UNITS) {
        h01[idx - (2 * UNITS * KXP + SEQT * NBATCH * KXP)] = (bf16_t)0.0f;  // buf 0
    } else if (idx < 2 * UNITS * KXP + SEQT * NBATCH * KXP + NBATCH * UNITS + NBATCH) {
        out_acc[idx - (2 * UNITS * KXP + SEQT * NBATCH * KXP + NBATCH * UNITS)] = 0.0f;
    } else if (idx == 2 * UNITS * KXP + SEQT * NBATCH * KXP + NBATCH * UNITS + NBATCH) {
        *bar = 0u;
    }
}

// ---------------------------------------------------------------------------
// Main persistent kernel: 64 WGs x 256 threads (8 waves). WG b owns columns
// [32*b, 32*b+32); wave w owns C tile (m = w&3, n-sub = w>>2).
// ---------------------------------------------------------------------------
__global__ __launch_bounds__(256, 2)
void rnn_persistent(const bf16_t* __restrict__ xbf,    // [T][B][96]
                    const bf16_t* __restrict__ Wx0t,   // [U][96]
                    const bf16_t* __restrict__ Wx5t,   // [U][96]
                    const bf16_t* __restrict__ Wh0t,   // [U][U]
                    const bf16_t* __restrict__ Wh5t,   // [U][U]
                    const float*  __restrict__ bvec,   // [6][U]
                    const float*  __restrict__ Wout,   // [U]
                    bf16_t* __restrict__ h01,          // [2][B][U] double buffer
                    float* __restrict__ out_acc, unsigned* __restrict__ bar)
{
    const int lane = threadIdx.x & 31;
    const int wid  = threadIdx.x >> 5;
    const int m0   = (wid & 3) * 16;                   // batch-row tile
    const int n0   = blockIdx.x * 32 + (wid >> 2) * 16;// output-column tile
    const int col  = lane & 15;
    const int lg   = lane >> 4;
    const int n    = n0 + col;                         // this lane's column
    const float b0 = bvec[n];
    const float b5 = bvec[5 * UNITS + n];

    __shared__ float lred[NBATCH];

    for (int t = 0; t < SEQT; ++t) {
        // integer-offset double buffering keeps GLOBAL address space
        const bf16_t* __restrict__ hprev = h01 + (size_t)(t & 1) * HBUF;
        bf16_t* __restrict__       hnext = h01 + (size_t)((t & 1) ^ 1) * HBUF;

        v8f c = { b0, b0, b0, b0, b0, b0, b0, b0 };
        // small GEMM: B fragments (24 VGPRs) intentionally hoisted across t
        c = wmma_gemm<KXP, false>(
                xbf + (size_t)t * NBATCH * KXP + (size_t)m0 * KXP, KXP,
                Wx0t + (size_t)n0 * KXP, KXP, c, lane);
        // big GEMM: opaque offsets keep loads inside the loop (L2-served)
        c = wmma_gemm<UNITS, true>(
                hprev + (size_t)m0 * UNITS, UNITS,
                Wh0t + (size_t)n0 * UNITS, UNITS, c, lane);

        // tanh, convert, scatter into next h buffer (C layout: row = 8*lg + v)
#pragma unroll
        for (int v = 0; v < 8; ++v) {
            int row = m0 + 8 * lg + v;
            hnext[(size_t)row * UNITS + n] = (bf16_t)tanhf(c[v]);
        }

        // ---- device-wide barrier between timesteps (monotonic counter) ----
        __threadfence();                               // release our stores
        __syncthreads();
        if (threadIdx.x == 0) {
            __hip_atomic_fetch_add(bar, 1u, __ATOMIC_RELAXED,
                                   __HIP_MEMORY_SCOPE_AGENT);
            const unsigned target = (unsigned)NBLK * (unsigned)(t + 1);
            while (__hip_atomic_load(bar, __ATOMIC_RELAXED,
                                     __HIP_MEMORY_SCOPE_AGENT) < target) {
                __builtin_amdgcn_s_sleep(8);
            }
        }
        __syncthreads();
        __threadfence();                               // acquire: drop stale lines
    }

    // ---- final layer 5 on h0[63] (lives in buffer 0 after 64 steps) ----
    const bf16_t* __restrict__ hfin = h01;             // t=63 wrote buffer 0
    v8f c = { b5, b5, b5, b5, b5, b5, b5, b5 };
    c = wmma_gemm<KXP, false>(
            xbf + (size_t)(SEQT - 1) * NBATCH * KXP + (size_t)m0 * KXP, KXP,
            Wx5t + (size_t)n0 * KXP, KXP, c, lane);
    c = wmma_gemm<UNITS, true>(
            hfin + (size_t)m0 * UNITS, UNITS,
            Wh5t + (size_t)n0 * UNITS, UNITS, c, lane);

    // out partial: sum over this WG's 32 columns of tanh(h5) * Wout[n]
    const float w = Wout[n];
    if (threadIdx.x < NBATCH) lred[threadIdx.x] = 0.0f;
    __syncthreads();
#pragma unroll
    for (int v = 0; v < 8; ++v) {
        int row = m0 + 8 * lg + v;
        atomicAdd(&lred[row], tanhf(c[v]) * w);
    }
    __syncthreads();
    if (threadIdx.x < NBATCH) {
        atomicAdd(&out_acc[threadIdx.x], lred[threadIdx.x]);
    }
}

// ---------------------------------------------------------------------------
// Finalize: sigmoid
// ---------------------------------------------------------------------------
__global__ void k_finalize(const float* __restrict__ acc,
                           const float* __restrict__ bout,
                           float* __restrict__ out)
{
    int i = threadIdx.x;
    if (i < NBATCH) {
        float x = acc[i] + bout[0];
        out[i] = 1.0f / (1.0f + expf(-x));
    }
}

// ---------------------------------------------------------------------------
// Launch
// ---------------------------------------------------------------------------
extern "C" void kernel_launch(void* const* d_in, const int* in_sizes, int n_in,
                              void* d_out, int out_size, void* d_ws, size_t ws_size,
                              hipStream_t stream)
{
    (void)in_sizes; (void)n_in; (void)out_size; (void)ws_size;

    const int*   inputs = (const int*)  d_in[0];   // [64][64] int32
    const float* emb    = (const float*)d_in[1];   // [8192][80]
    const float* Wx     = (const float*)d_in[2];   // [6][80][2048]
    const float* Wh     = (const float*)d_in[3];   // [6][2048][2048]
    const float* bvec   = (const float*)d_in[4];   // [6][2048]
    const float* Wout   = (const float*)d_in[5];   // [2048][1]
    const float* bout   = (const float*)d_in[6];   // [1]
    float* out = (float*)d_out;                    // [64]

    // ---- workspace carve (all offsets multiples of 256 bytes) ----
    char* ws = (char*)d_ws;
    size_t off = 0;
    bf16_t* Wh0t = (bf16_t*)(ws + off); off += (size_t)UNITS * UNITS * 2;        // 8 MB
    bf16_t* Wh5t = (bf16_t*)(ws + off); off += (size_t)UNITS * UNITS * 2;        // 8 MB
    bf16_t* Wx0t = (bf16_t*)(ws + off); off += (size_t)UNITS * KXP * 2;          // 384 KB
    bf16_t* Wx5t = (bf16_t*)(ws + off); off += (size_t)UNITS * KXP * 2;          // 384 KB
    bf16_t* xbf  = (bf16_t*)(ws + off); off += (size_t)SEQT * NBATCH * KXP * 2;  // 768 KB
    bf16_t* h01  = (bf16_t*)(ws + off); off += 2 * HBUF * 2;                     // 512 KB
    float*  oacc = (float*) (ws + off); off += 256;
    unsigned* bar = (unsigned*)(ws + off); off += 256;

    // 1) weight transpose/convert: 2 * 2048 * 2048 elements
    {
        int total = 2 * UNITS * UNITS;
        k_transpose_wh<<<(total + 255) / 256, 256, 0, stream>>>(Wh, Wh0t, Wh5t);
    }
    // 2) Wx transpose + embedding gather + zeroing
    {
        int total = 2 * UNITS * KXP + SEQT * NBATCH * KXP + NBATCH * UNITS + NBATCH + 1;
        k_prep_misc<<<(total + 255) / 256, 256, 0, stream>>>(
            Wx, emb, inputs, Wx0t, Wx5t, xbf, h01, oacc, bar);
    }
    // 3) persistent RNN (64 co-resident WGs, device barrier per timestep)
    rnn_persistent<<<NBLK, 256, 0, stream>>>(
        xbf, Wx0t, Wx5t, Wh0t, Wh5t, bvec, Wout, h01, oacc, bar);
    // 4) sigmoid
    k_finalize<<<1, 64, 0, stream>>>(oacc, bout, out);
}
